// CausalSelfAttentionSynapse_52329881535149
// MI455X (gfx1250) — matrix operand order repaired and checked by reading, not compiled, not run on hardware
//
#include <hip/hip_runtime.h>
#include <math.h>

#define B_ 4
#define T_ 2048
#define E_ 1024
#define H_ 16
#define D_ 64
// Lift GEMM: M = B_*T_ = 8192, N = 2*E_ = 2048, K = E_ = 1024

typedef float  v2f   __attribute__((ext_vector_type(2)));
typedef float  v8f   __attribute__((ext_vector_type(8)));
typedef __bf16 v4bf  __attribute__((ext_vector_type(4)));
typedef __bf16 v8bf  __attribute__((ext_vector_type(8)));
typedef __bf16 v16bf __attribute__((ext_vector_type(16)));

#define WMMA_BF16(A, Bm, C)                                                 \
  __builtin_amdgcn_wmma_f32_16x16x32_bf16(false, (A), false, (Bm),          \
                                          (short)0, (C), false, false)
#define WMMA_F32(A, Bm, C)                                                  \
  __builtin_amdgcn_wmma_f32_16x16x4_f32(false, (A), false, (Bm),            \
                                        (short)0, (C), false, false)

// ---------------------------------------------------------------------------
// Kernel 0: split-precision conversion  x(f32) -> hi(bf16) + lo(bf16 residual)
// a = hi + lo with |lo| <= 2^-8 |a|; dropped lo*lo term => ~2^-16 rel error.
// ---------------------------------------------------------------------------
__global__ __launch_bounds__(256) void split_bf16_kernel(
    const float* __restrict__ x, __bf16* __restrict__ hi,
    __bf16* __restrict__ lo, int n4) {
  const int i = blockIdx.x * 256 + threadIdx.x;
  if (i >= n4) return;
  const float4 v = ((const float4*)x)[i];
  const float f[4] = {v.x, v.y, v.z, v.w};
  v4bf h, l;
  #pragma unroll
  for (int j = 0; j < 4; ++j) {
    const __bf16 hj = (__bf16)f[j];
    h[j] = hj;
    l[j] = (__bf16)(f[j] - (float)hj);
  }
  *(v4bf*)(hi + 4 * (size_t)i) = h;
  *(v4bf*)(lo + 4 * (size_t)i) = l;
}

// ---------------------------------------------------------------------------
// Kernel 1: qk = g @ W_lift^T with split-bf16 V_WMMA_F32_16X16X32_BF16.
// acc += Ahi*Bhi + Ahi*Blo + Alo*Bhi (f32 accumulate) ~ f32 GEMM accuracy.
// Block: 256 threads = 8 waves (4 M-slices x 2 N-slices), tile 128x64,
// K-tile = 64 bf16. LDS row stride 72 bf16: 16B-aligned chunks, the 16
// fragment reads/step touch all 64 banks exactly once.
// Epilogue scatters f32 results into split q/k [B,H,T,D] workspaces.
// ---------------------------------------------------------------------------
__device__ inline v16bf load_a_frag(const __bf16* row, int kb, int half) {
  // A(16x32) lane layout: chunk0 K = kb+8h..+7, chunk1 K = kb+16+8h..+7
  const v8bf c0 = *(const v8bf*)(row + kb + 8 * half);
  const v8bf c1 = *(const v8bf*)(row + kb + 16 + 8 * half);
  return __builtin_shufflevector(c0, c1, 0, 1, 2, 3, 4, 5, 6, 7, 8, 9, 10, 11,
                                 12, 13, 14, 15);
}
__device__ inline v16bf load_b_frag(const __bf16* row, int kb, int half) {
  // B(32x16) lane layout: chunk0 K = kb+8h..+7, chunk1 K = kb+16+8h..+7 of column n (= W row n)
  const v8bf c0 = *(const v8bf*)(row + kb + 8 * half);
  const v8bf c1 = *(const v8bf*)(row + kb + 16 + 8 * half);
  return __builtin_shufflevector(c0, c1, 0, 1, 2, 3, 4, 5, 6, 7, 8, 9, 10, 11,
                                 12, 13, 14, 15);
}

__global__ __launch_bounds__(256) void lift_gemm_bf16_kernel(
    const __bf16* __restrict__ ghi, const __bf16* __restrict__ glo,
    const __bf16* __restrict__ whi, const __bf16* __restrict__ wlo,
    float* __restrict__ qws, float* __restrict__ kws) {
  __shared__ __bf16 Ahi[128][72], Alo[128][72];
  __shared__ __bf16 Bhi[64][72], Blo[64][72];

  const int tid  = threadIdx.x;
  const int lane = tid & 31;
  const int wave = tid >> 5;
  const int wm   = wave & 3;   // 32-row slice
  const int wn   = wave >> 2;  // 32-col slice
  const int m0   = blockIdx.x * 128;
  const int n0   = blockIdx.y * 64;
  const int half = lane >> 4;
  const int l16  = lane & 15;

  v8f acc[2][2];
  acc[0][0] = {}; acc[0][1] = {}; acc[1][0] = {}; acc[1][1] = {};

  const __bf16* arow0h = &Ahi[wm * 32 + l16][0];
  const __bf16* arow1h = &Ahi[wm * 32 + 16 + l16][0];
  const __bf16* arow0l = &Alo[wm * 32 + l16][0];
  const __bf16* arow1l = &Alo[wm * 32 + 16 + l16][0];
  const __bf16* brow0h = &Bhi[wn * 32 + l16][0];
  const __bf16* brow1h = &Bhi[wn * 32 + 16 + l16][0];
  const __bf16* brow0l = &Blo[wn * 32 + l16][0];
  const __bf16* brow1l = &Blo[wn * 32 + 16 + l16][0];

  for (int k0 = 0; k0 < E_; k0 += 64) {
    // Cooperative tile copies: 16B (8 x bf16) chunks, fully coalesced.
    #pragma unroll
    for (int i = tid; i < 128 * 8; i += 256) {
      const int r = i >> 3, c = (i & 7) << 3;
      const size_t src = (size_t)(m0 + r) * E_ + k0 + c;
      *(uint4*)&Ahi[r][c] = *(const uint4*)(ghi + src);
      *(uint4*)&Alo[r][c] = *(const uint4*)(glo + src);
    }
    #pragma unroll
    for (int i = tid; i < 64 * 8; i += 256) {
      const int r = i >> 3, c = (i & 7) << 3;
      const size_t src = (size_t)(n0 + r) * E_ + k0 + c;
      *(uint4*)&Bhi[r][c] = *(const uint4*)(whi + src);
      *(uint4*)&Blo[r][c] = *(const uint4*)(wlo + src);
    }
    __syncthreads();

    #pragma unroll
    for (int kb = 0; kb < 64; kb += 32) {
      const v16bf a0h = load_a_frag(arow0h, kb, half);
      const v16bf a1h = load_a_frag(arow1h, kb, half);
      const v16bf a0l = load_a_frag(arow0l, kb, half);
      const v16bf a1l = load_a_frag(arow1l, kb, half);
      const v16bf b0h = load_b_frag(brow0h, kb, half);
      const v16bf b1h = load_b_frag(brow1h, kb, half);
      const v16bf b0l = load_b_frag(brow0l, kb, half);
      const v16bf b1l = load_b_frag(brow1l, kb, half);

      acc[0][0] = WMMA_BF16(a0h, b0h, acc[0][0]);
      acc[0][1] = WMMA_BF16(a0h, b1h, acc[0][1]);
      acc[1][0] = WMMA_BF16(a1h, b0h, acc[1][0]);
      acc[1][1] = WMMA_BF16(a1h, b1h, acc[1][1]);

      acc[0][0] = WMMA_BF16(a0h, b0l, acc[0][0]);
      acc[0][1] = WMMA_BF16(a0h, b1l, acc[0][1]);
      acc[1][0] = WMMA_BF16(a1h, b0l, acc[1][0]);
      acc[1][1] = WMMA_BF16(a1h, b1l, acc[1][1]);

      acc[0][0] = WMMA_BF16(a0l, b0h, acc[0][0]);
      acc[0][1] = WMMA_BF16(a0l, b1h, acc[0][1]);
      acc[1][0] = WMMA_BF16(a1l, b0h, acc[1][0]);
      acc[1][1] = WMMA_BF16(a1l, b1h, acc[1][1]);
    }
    __syncthreads();
  }

  // C/D layout: lane -> col n = l16, VGPR r -> row M = r + 8*half.
  #pragma unroll
  for (int i = 0; i < 2; ++i) {
    #pragma unroll
    for (int j = 0; j < 2; ++j) {
      const int n = n0 + wn * 32 + j * 16 + l16;
      #pragma unroll
      for (int r = 0; r < 8; ++r) {
        const int m = m0 + wm * 32 + i * 16 + r + 8 * half;
        const int b = m >> 11;
        const int t = m & (T_ - 1);
        const float val = acc[i][j][r];
        if (n < E_) {
          const int h = n >> 6, d = n & 63;
          qws[((((size_t)b * H_ + h) * T_ + t) << 6) + d] = val;
        } else {
          const int n2 = n - E_;
          const int h = n2 >> 6, d = n2 & 63;
          kws[((((size_t)b * H_ + h) * T_ + t) << 6) + d] = val;
        }
      }
    }
  }
}

// ---------------------------------------------------------------------------
// Kernel 2: causal logsumexp of S = Q K^T per (b,h) with f32 WMMA (exp-
// sensitive + L2-bandwidth bound, so full f32 fidelity kept here).
// One wave per 16-query tile; Q fragments register-resident; per key tile
// 16 chained V_WMMA_F32_16X16X4_F32, then online max/sum via 16-lane
// butterfly shuffles.
// ---------------------------------------------------------------------------
__global__ __launch_bounds__(32) void attn_lse_kernel(
    const float* __restrict__ qws, const float* __restrict__ kws,
    float* __restrict__ lse) {
  const int qt = blockIdx.x;
  const int bh = blockIdx.y;
  const int lane = threadIdx.x;
  const int half = lane >> 4;
  const int l16  = lane & 15;

  const float* qbase = qws + (size_t)bh * T_ * D_;
  const float* kbase = kws + (size_t)bh * T_ * D_;

  v2f qa[16];
  const float* qrow = qbase + ((size_t)(qt * 16 + l16) << 6);
  #pragma unroll
  for (int kk = 0; kk < 16; ++kk)
    qa[kk] = *(const v2f*)(qrow + kk * 4 + 2 * half);

  float mrow[8], srow[8];
  #pragma unroll
  for (int r = 0; r < 8; ++r) { mrow[r] = -INFINITY; srow[r] = 0.0f; }

  for (int kt = 0; kt <= qt; ++kt) {
    const float* krow = kbase + ((size_t)(kt * 16 + l16) << 6);
    v8f s = {};
    #pragma unroll
    for (int kk = 0; kk < 16; ++kk) {
      const v2f bf = *(const v2f*)(krow + kk * 4 + 2 * half);
      s = WMMA_F32(qa[kk], bf, s);
    }

    const int kg = kt * 16 + l16;
    #pragma unroll
    for (int r = 0; r < 8; ++r) {
      const int qg = qt * 16 + r + 8 * half;
      float v = s[r];
      if (kg > qg) v = -INFINITY;  // causal mask
      float vmax = v;
      vmax = fmaxf(vmax, __shfl_xor(vmax, 1, 16));
      vmax = fmaxf(vmax, __shfl_xor(vmax, 2, 16));
      vmax = fmaxf(vmax, __shfl_xor(vmax, 4, 16));
      vmax = fmaxf(vmax, __shfl_xor(vmax, 8, 16));
      const float mnew = fmaxf(mrow[r], vmax);  // diag always unmasked
      float e = expf(v - mnew);                 // -inf -> 0
      e += __shfl_xor(e, 1, 16);
      e += __shfl_xor(e, 2, 16);
      e += __shfl_xor(e, 4, 16);
      e += __shfl_xor(e, 8, 16);
      srow[r] = srow[r] * expf(mrow[r] - mnew) + e;
      mrow[r] = mnew;
    }
  }

  #pragma unroll
  for (int r = 0; r < 8; ++r) {
    if (l16 == r) {
      lse[(size_t)bh * T_ + qt * 16 + r + 8 * half] = mrow[r] + logf(srow[r]);
    }
  }
}

// ---------------------------------------------------------------------------
// Kernel 3: out[b,t] = sum_h lse[b,h,t] * colsum_g(W_proj[g,h])
// ---------------------------------------------------------------------------
__global__ __launch_bounds__(256) void proj_out_kernel(
    const float* __restrict__ lse, const float* __restrict__ Wp,
    float* __restrict__ out) {
  __shared__ float cs[H_];
  if (threadIdx.x < H_) {
    float sum = 0.0f;
    #pragma unroll
    for (int gg = 0; gg < H_; ++gg) sum += Wp[gg * H_ + threadIdx.x];
    cs[threadIdx.x] = sum;
  }
  __syncthreads();

  const int idx = blockIdx.x * 256 + threadIdx.x;
  const int b = idx >> 11;
  const int t = idx & (T_ - 1);
  float accv = 0.0f;
  #pragma unroll
  for (int h = 0; h < H_; ++h)
    accv += cs[h] * lse[(((size_t)b * H_ + h) * T_) + t];
  out[idx] = accv;
}

// ---------------------------------------------------------------------------
extern "C" void kernel_launch(void* const* d_in, const int* in_sizes, int n_in,
                              void* d_out, int out_size, void* d_ws,
                              size_t ws_size, hipStream_t stream) {
  (void)in_sizes; (void)n_in; (void)out_size; (void)ws_size;
  const float* g      = (const float*)d_in[0];  // (B,T,E)
  const float* W_lift = (const float*)d_in[1];  // (2E,E)
  const float* W_proj = (const float*)d_in[2];  // (H,H)
  float* out = (float*)d_out;                   // (B*T)

  // Workspace: q[B,H,T,D] f32 | k[...] f32 | lse[B,H,T] f32 |
  //            ghi,glo bf16[B*T*E] | whi,wlo bf16[2E*E]   (~106 MB total)
  const size_t NQK = (size_t)B_ * H_ * T_ * D_;  // 8388608
  const size_t NG  = (size_t)B_ * T_ * E_;       // 8388608
  const size_t NW  = (size_t)2 * E_ * E_;        // 2097152
  float*  qws = (float*)d_ws;
  float*  kws = qws + NQK;
  float*  lse = kws + NQK;
  __bf16* ghi = (__bf16*)(lse + (size_t)B_ * H_ * T_);
  __bf16* glo = ghi + NG;
  __bf16* whi = glo + NG;
  __bf16* wlo = whi + NW;

  split_bf16_kernel<<<(int)(NG / 4 / 256), 256, 0, stream>>>(g, ghi, glo,
                                                             (int)(NG / 4));
  split_bf16_kernel<<<(int)(NW / 4 / 256), 256, 0, stream>>>(W_lift, whi, wlo,
                                                             (int)(NW / 4));

  dim3 grid1((B_ * T_) / 128, (2 * E_) / 64);  // 64 x 32
  lift_gemm_bf16_kernel<<<grid1, 256, 0, stream>>>(ghi, glo, whi, wlo, qws,
                                                   kws);

  dim3 grid2(T_ / 16, B_ * H_);                // 128 x 64
  attn_lse_kernel<<<grid2, 32, 0, stream>>>(qws, kws, lse);

  proj_out_kernel<<<(B_ * T_) / 256, 256, 0, stream>>>(lse, W_proj, out);
}
